// TempUpdater_14534169330112
// MI455X (gfx1250) — matrix operand order repaired and checked by reading, not compiled, not run on hardware
//
#include <hip/hip_runtime.h>
#include <hip/hip_bf16.h>

// ---------------------------------------------------------------------------
// MI455X (gfx1250): all large matmuls via v_wmma_f32_16x16x32_bf16.
// B=32, C=256, N=1024, H=2048.  All GEMMs are "NT": A row-major [M,K] bf16,
// B stored as [N,K] bf16 (computes A x B^T), so every WMMA fragment is a pair
// of contiguous 16B K-runs for both operands.
// Tile staging uses GLOBAL_LOAD_ASYNC_TO_LDS_B128 (ASYNCcnt) with a
// triple-buffered LDS pipeline: one workgroup barrier per K-step.
// ---------------------------------------------------------------------------

typedef unsigned short u16;
typedef __attribute__((ext_vector_type(16))) __bf16 v16bf;
typedef __attribute__((ext_vector_type(8)))  float  v8f;
typedef __attribute__((ext_vector_type(8)))  unsigned short u16x8;
typedef __attribute__((ext_vector_type(4)))  unsigned short u16x4;
typedef int v4i __attribute__((vector_size(16)));   // matches builtin prototype

union FragBF { u16x8 u[2]; v16bf v; };
static_assert(sizeof(FragBF) == 32, "frag size");

__device__ __forceinline__ u16 f2bf(float f) {
  unsigned int u = __float_as_uint(f);
  u += 0x7FFFu + ((u >> 16) & 1u);   // round-to-nearest-even
  return (u16)(u >> 16);
}
__device__ __forceinline__ float bf2f(u16 x) {
  return __uint_as_float(((unsigned int)x) << 16);
}

// ---- async global->LDS 16B copy (gfx1250 ASYNCcnt path) -------------------
// LDS generic pointers keep the byte offset in addr[31:0] (LDS aperture), so
// integer round-trips produce valid AS(1)/AS(3) pointers.
__device__ __forceinline__ void async_copy16(const void* g, void* l) {
#if __has_builtin(__builtin_amdgcn_global_load_async_to_lds_b128)
  __builtin_amdgcn_global_load_async_to_lds_b128(
      (__attribute__((address_space(1))) v4i*)(uintptr_t)g,
      (__attribute__((address_space(3))) v4i*)(uintptr_t)l,
      0, 0);
#else
  unsigned int loff = (unsigned int)(uintptr_t)l;
  asm volatile("global_load_async_to_lds_b128 %0, %1, off"
               :: "v"(loff), "v"(g) : "memory");
#endif
}

#if __has_builtin(__builtin_amdgcn_s_wait_asynccnt)
#define WAIT_ASYNC(n) __builtin_amdgcn_s_wait_asynccnt(n)
#else
#define WAIT_ASYNC(n) asm volatile("s_wait_asynccnt " #n ::: "memory")
#endif

// ---------------------------------------------------------------------------
// Elementwise fp32 -> bf16 convert (vectorized x4)
// ---------------------------------------------------------------------------
__global__ __launch_bounds__(256)
void f32_to_bf16_vec(const float* __restrict__ s, u16* __restrict__ d, int n4) {
  int i = blockIdx.x * 256 + threadIdx.x;
  if (i < n4) {
    float4 v = ((const float4*)s)[i];
    u16x4 o;
    o[0] = f2bf(v.x); o[1] = f2bf(v.y); o[2] = f2bf(v.z); o[3] = f2bf(v.w);
    ((u16x4*)d)[i] = o;
  }
}

// ---------------------------------------------------------------------------
// [C,N] fp32 -> [N,C] (bf16, and optionally fp32) per batch, 32x32 LDS tiles
// ---------------------------------------------------------------------------
template<int WRITE_F32>
__global__ __launch_bounds__(256)
void transpose_cn_to_nc(const float* __restrict__ src, u16* __restrict__ dst_bf,
                        float* __restrict__ dst_f32,
                        long long sStride, long long dStride) {
  __shared__ float tile[32][33];
  const int b = blockIdx.z;
  const int n0 = blockIdx.x * 32, c0 = blockIdx.y * 32;
  const int tx = threadIdx.x & 31, ty = threadIdx.x >> 5; // ty: 0..7
  const float* s = src + (size_t)b * sStride;
  for (int r = 0; r < 4; ++r) {
    int ci = ty + r * 8;
    tile[ci][tx] = s[(size_t)(c0 + ci) * 1024 + n0 + tx];
  }
  __syncthreads();
  for (int r = 0; r < 4; ++r) {
    int ni = ty + r * 8;
    float v = tile[tx][ni];
    size_t di = (size_t)b * dStride + (size_t)(n0 + ni) * 256 + c0 + tx;
    dst_bf[di] = f2bf(v);
    if (WRITE_F32) dst_f32[di] = v;
  }
}

// ---------------------------------------------------------------------------
// Row softmax over 1024 elements, with 1/8 logit scale, bf16 output
// ---------------------------------------------------------------------------
__global__ __launch_bounds__(256)
void softmax_rows(const float* __restrict__ S, u16* __restrict__ att) {
  __shared__ float red[256];
  const int row = blockIdx.x;
  const int t = threadIdx.x;
  float4 v = ((const float4*)(S + (size_t)row * 1024))[t];
  float a = v.x * 0.125f, b = v.y * 0.125f, c = v.z * 0.125f, d = v.w * 0.125f;
  float m = fmaxf(fmaxf(a, b), fmaxf(c, d));
  red[t] = m; __syncthreads();
  for (int s = 128; s > 0; s >>= 1) {
    if (t < s) red[t] = fmaxf(red[t], red[t + s]);
    __syncthreads();
  }
  m = red[0]; __syncthreads();
  float ea = __expf(a - m), eb = __expf(b - m), ec = __expf(c - m), ed = __expf(d - m);
  red[t] = ea + eb + ec + ed; __syncthreads();
  for (int s = 128; s > 0; s >>= 1) {
    if (t < s) red[t] += red[t + s];
    __syncthreads();
  }
  float inv = 1.0f / red[0];
  u16x4 o;
  o[0] = f2bf(ea * inv); o[1] = f2bf(eb * inv);
  o[2] = f2bf(ec * inv); o[3] = f2bf(ed * inv);
  ((u16x4*)(att + (size_t)row * 1024))[t] = o;
}

// ---------------------------------------------------------------------------
// NT GEMM: D[M,Nn] = A[M,K] x Bm[Nn,K]^T,  bf16 in, fp32 accumulate (WMMA).
// Block tile 128x128, BK=32, 8 waves, each wave 64x32 (4x2 WMMA accumulators).
// Triple-buffered LDS fed by async global->LDS b128 copies; one barrier/step.
// MODE 0: store fp32
// MODE 1: += bf16 residual (resid[gm*ldr+gn]); store fp32 (Df) and bf16 (Dbf)
// MODE 2: += bias[gn]; relu; store bf16
// MODE 3: += bias[gn]; store fp32
// ---------------------------------------------------------------------------
#define GPAD 40   // LDS row pitch in u16 (32 data + 8 pad), keeps 16B alignment

template<int MODE>
__global__ __launch_bounds__(256)
void gemm_nt_bf16(const u16* __restrict__ A, int lda, long long strideA,
                  const u16* __restrict__ Bm, int ldb, long long strideB,
                  float* __restrict__ Df, u16* __restrict__ Dbf,
                  int ldd, long long strideD,
                  const float* __restrict__ bias,
                  const u16* __restrict__ resid, long long strideR, int ldr,
                  int K) {
  __shared__ u16 smem[3][2 * 128 * GPAD];

  const int tid  = threadIdx.x;
  const int lane = tid & 31;
  const int wave = tid >> 5;
  const int z    = blockIdx.z;
  const int m0   = blockIdx.y * 128;
  const int n0   = blockIdx.x * 128;

  const u16* Ab = A  + (size_t)z * strideA;
  const u16* Bb = Bm + (size_t)z * strideB;

  const int R0 = (wave >> 2) * 64;   // wave row origin in block tile
  const int C0 = (wave & 3) * 32;    // wave col origin
  const int h  = lane >> 4;          // half-wave
  const int r  = lane & 15;

  v8f acc[4][2];
  for (int s = 0; s < 4; ++s)
    for (int t = 0; t < 2; ++t)
      for (int i = 0; i < 8; ++i) acc[s][t][i] = 0.0f;

  // staging map: 256 threads, 2 threads/row, 32B each (two async b128s)
  const int rowg  = tid >> 1;
  const int halfg = tid & 1;
  const u16* pA = Ab + (size_t)(m0 + rowg) * lda + halfg * 16;
  const u16* pB = Bb + (size_t)(n0 + rowg) * ldb + halfg * 16;
  const int  soff = rowg * GPAD + halfg * 16;

  const int KT = K >> 5;

  auto issue = [&](int buf, int kt) {
    const u16* qA = pA + (size_t)kt * 32;
    u16* dA = &smem[buf][soff];
    async_copy16(qA,     dA);
    async_copy16(qA + 8, dA + 8);
    const u16* qB = pB + (size_t)kt * 32;
    u16* dB = &smem[buf][128 * GPAD + soff];
    async_copy16(qB,     dB);
    async_copy16(qB + 8, dB + 8);
  };

  issue(0, 0);
  if (KT > 1) issue(1, 1);

  for (int kt = 0; kt < KT; ++kt) {
    const int cur = kt % 3;
    // retire the current tile's batch (4 ops); in-order completion means
    // leaving <=4 outstanding retires everything older than the newest batch
    if (kt + 1 < KT) { WAIT_ASYNC(4); } else { WAIT_ASYNC(0); }
    __syncthreads();                      // make all waves' tile data visible
    if (kt + 2 < KT) issue((kt + 2) % 3, kt + 2);  // safe: buffer last read at
                                                   // kt-1, fenced by barrier

    const u16* sA = &smem[cur][0];
    const u16* sB = &smem[cur][128 * GPAD];
    FragBF fa[4], fb[2];
    // A fragment: lane holds row (R+r), K runs {8h..8h+7} and {16+8h..23+8h}
    for (int s = 0; s < 4; ++s) {
      const int base = (R0 + s * 16 + r) * GPAD + 8 * h;
      fa[s].u[0] = *(const u16x8*)(sA + base);
      fa[s].u[1] = *(const u16x8*)(sA + base + 16);
    }
    // B fragment: lane holds col (C+r), K runs {8h..8h+7} and {16+8h..23+8h}
    for (int t = 0; t < 2; ++t) {
      const int base = (C0 + t * 16 + r) * GPAD + 8 * h;
      fb[t].u[0] = *(const u16x8*)(sB + base);
      fb[t].u[1] = *(const u16x8*)(sB + base + 16);
    }
    for (int s = 0; s < 4; ++s)
      for (int t = 0; t < 2; ++t)
        acc[s][t] = __builtin_amdgcn_wmma_f32_16x16x32_bf16(
            false, fa[s].v, false, fb[t].v, (short)0, acc[s][t], false, false);
  }

  // Epilogue. D f32 layout: VGPR i -> (M = i + 8h, N = lane&15)
  for (int s = 0; s < 4; ++s) {
    for (int t = 0; t < 2; ++t) {
      const int gmB = m0 + R0 + s * 16 + 8 * h;
      const int gn  = n0 + C0 + t * 16 + r;
      for (int i = 0; i < 8; ++i) {
        const int gm = gmB + i;
        float v = acc[s][t][i];
        if (MODE == 0) {
          Df[(size_t)z * strideD + (size_t)gm * ldd + gn] = v;
        } else if (MODE == 1) {
          v += bf2f(resid[(size_t)z * strideR + (size_t)gm * ldr + gn]);
          size_t di = (size_t)z * strideD + (size_t)gm * ldd + gn;
          Df[di]  = v;
          Dbf[di] = f2bf(v);
        } else if (MODE == 2) {
          v += bias[gn];
          v = fmaxf(v, 0.0f);
          Dbf[(size_t)gm * ldd + gn] = f2bf(v);
        } else {
          v += bias[gn];
          Df[(size_t)gm * ldd + gn] = v;
        }
      }
    }
  }
}

// ---------------------------------------------------------------------------
// out[b,c,n] = o_pre[b,n,c] + t_f32[b,n,c]  (coalesced LDS tile transpose)
// ---------------------------------------------------------------------------
__global__ __launch_bounds__(256)
void add_transpose_out(const float* __restrict__ opre, const float* __restrict__ tres,
                       float* __restrict__ out) {
  __shared__ float tile[32][33];
  const int b = blockIdx.z;
  const int n0 = blockIdx.x * 32, c0 = blockIdx.y * 32;
  const int tx = threadIdx.x & 31, ty = threadIdx.x >> 5;
  const size_t base = (size_t)b * 2048 * 256;
  for (int r = 0; r < 4; ++r) {
    int ni = ty + r * 8;
    size_t idx = base + (size_t)(n0 + ni) * 256 + c0 + tx;
    tile[ni][tx] = opre[idx] + tres[idx];
  }
  __syncthreads();
  const size_t obase = (size_t)b * 256 * 2048;
  for (int r = 0; r < 4; ++r) {
    int ci = ty + r * 8;
    out[obase + (size_t)(c0 + ci) * 2048 + n0 + tx] = tile[tx][ci];
  }
}

// ---------------------------------------------------------------------------
extern "C" void kernel_launch(void* const* d_in, const int* in_sizes, int n_in,
                              void* d_out, int out_size, void* d_ws, size_t ws_size,
                              hipStream_t stream) {
  constexpr int Bc = 32, Cc = 256, Nc = 1024, Hc = 2048, N2 = 2048;

  const float* t0 = (const float*)d_in[0];
  const float* t1 = (const float*)d_in[1];
  const float* W1 = (const float*)d_in[2];
  const float* b1 = (const float*)d_in[3];
  const float* W2 = (const float*)d_in[4];
  const float* b2 = (const float*)d_in[5];
  float* out = (float*)d_out;

  char* w = (char*)d_ws;
  size_t off = 0;
  auto alloc = [&](size_t bytes) -> void* {
    void* p = w + off;
    off += (bytes + 255) & ~(size_t)255;
    return p;
  };
  u16*   t0_bf  = (u16*)  alloc((size_t)Bc * Cc * Nc * 2);  // t0 as bf16 [B,C,N]
  u16*   t1t_bf = (u16*)  alloc((size_t)Bc * Nc * Cc * 2);  // t1^T bf16 [B,N,C]
  u16*   t_bf   = (u16*)  alloc((size_t)Bc * N2 * Cc * 2);  // concat t, bf16 [B,2N,C]
  float* t_f32  = (float*)alloc((size_t)Bc * N2 * Cc * 4);  // concat t, fp32 (residual)
  float* Sbuf   = (float*)alloc((size_t)Bc * Nc * Nc * 4);  // logits; later reused as o_pre
  u16*   att_bf = (u16*)  alloc((size_t)Bc * Nc * Nc * 2);  // softmax probs bf16
  u16*   h_bf   = (u16*)  alloc((size_t)Bc * N2 * Hc * 2);  // hidden bf16
  u16*   W1_bf  = (u16*)  alloc((size_t)Hc * Cc * 2);
  u16*   W2_bf  = (u16*)  alloc((size_t)Cc * Hc * 2);
  float* o_pre  = Sbuf;   // reuse logits buffer (67MB needed <= 134MB)

  // --- converts ---
  { int n4 = Bc * Cc * Nc / 4; f32_to_bf16_vec<<<(n4 + 255) / 256, 256, 0, stream>>>(t0, t0_bf, n4); }
  { int n4 = Hc * Cc / 4;      f32_to_bf16_vec<<<(n4 + 255) / 256, 256, 0, stream>>>(W1, W1_bf, n4); }
  { int n4 = Cc * Hc / 4;      f32_to_bf16_vec<<<(n4 + 255) / 256, 256, 0, stream>>>(W2, W2_bf, n4); }

  // --- transposes: t0 -> t (first half, bf16+fp32), t1 -> t1^T bf16 ---
  dim3 tg(Nc / 32, Cc / 32, Bc);
  transpose_cn_to_nc<1><<<tg, 256, 0, stream>>>(t0, t_bf, t_f32,
                                                (long long)Cc * Nc, (long long)N2 * Cc);
  transpose_cn_to_nc<0><<<tg, 256, 0, stream>>>(t1, t1t_bf, nullptr,
                                                (long long)Cc * Nc, (long long)Nc * Cc);

  // --- Stage 1: S = t1^T x t0  (per batch, M=N=1024, K=C=256) ---
  gemm_nt_bf16<0><<<dim3(Nc / 128, Nc / 128, Bc), 256, 0, stream>>>(
      t1t_bf, Cc, (long long)Nc * Cc,
      t_bf,   Cc, (long long)N2 * Cc,          // first half of t = t0^T
      Sbuf, nullptr, Nc, (long long)Nc * Nc,
      nullptr, nullptr, 0, 0, Cc);

  // --- softmax(S/8) -> att (bf16) ---
  softmax_rows<<<dim3(Bc * Nc), 256, 0, stream>>>(Sbuf, att_bf);

  // --- Stage 2: t1' = att x t0^T + t1^T  (M=1024, Nn=C=256, K=1024) ---
  gemm_nt_bf16<1><<<dim3(Cc / 128, Nc / 128, Bc), 256, 0, stream>>>(
      att_bf, Nc, (long long)Nc * Nc,
      t0_bf,  Nc, (long long)Cc * Nc,          // B[n=c][k] = t0[c,k]
      t_f32 + (size_t)Nc * Cc, t_bf + (size_t)Nc * Cc, Cc, (long long)N2 * Cc,
      nullptr, t1t_bf, (long long)Nc * Cc, Cc, Nc);

  // --- Stage 3: h = relu(t x W1^T + b1)  (M=B*2N=65536, Nn=H, K=C) ---
  gemm_nt_bf16<2><<<dim3(Hc / 128, (Bc * N2) / 128, 1), 256, 0, stream>>>(
      t_bf,  Cc, 0,
      W1_bf, Cc, 0,
      nullptr, h_bf, Hc, 0,
      b1, nullptr, 0, 0, Cc);

  // --- Stage 4: o_pre = h x W2^T + b2  (M=65536, Nn=C, K=H) ---
  gemm_nt_bf16<3><<<dim3(Cc / 128, (Bc * N2) / 128, 1), 256, 0, stream>>>(
      h_bf,  Hc, 0,
      W2_bf, Hc, 0,
      o_pre, nullptr, Cc, 0,
      b2, nullptr, 0, 0, Hc);

  // --- final: out[b,c,n] = o_pre[b,n,c] + t[b,n,c]  (coalesced transpose) ---
  add_transpose_out<<<dim3(N2 / 32, Cc / 32, Bc), 256, 0, stream>>>(o_pre, t_f32, out);
}
